// MultiHeadAttention_58385785422211
// MI455X (gfx1250) — matrix operand order repaired and checked by reading, not compiled, not run on hardware
//
#include <hip/hip_runtime.h>
#include <hip/hip_bf16.h>

// Problem constants (fixed by the reference).
#define B_   2
#define S_   2048
#define D_   1024
#define H_   16
#define HD_  64
#define M_   (B_ * S_)          // 4096 rows for all projection GEMMs

typedef __attribute__((ext_vector_type(8)))  unsigned short us8;
typedef __attribute__((ext_vector_type(16))) __bf16         v16bf;
typedef __attribute__((ext_vector_type(8)))  float          v8f;

union FragU { us8 h[2]; v16bf f; };

__device__ __forceinline__ unsigned short f2bf(float f) {
  unsigned int u = __float_as_uint(f);
  u += 0x7FFFu + ((u >> 16) & 1u);   // round-to-nearest-even
  return (unsigned short)(u >> 16);
}

__device__ __forceinline__ v8f wmma_bf16(v16bf a, v16bf b, v8f c) {
  return __builtin_amdgcn_wmma_f32_16x16x32_bf16(
      /*neg_a=*/false, a, /*neg_b=*/false, b,
      /*c_mod=*/(short)0, c, /*reuse_a=*/false, /*reuse_b=*/false);
}

// A-operand fragment, 16x32 bf16, row-major source with leading dim `ld`.
// ISA layout: lanes 0-15 row M=lane, K = {0..7} u {16..23}; lanes 16-31 row
// M=lane-16, K = {8..15} u {24..31}.  -> two 16B loads per lane.
__device__ __forceinline__ v16bf load_fragA(const unsigned short* base, int ld,
                                            int row0, int k0, int lane) {
  const int half = (lane >> 4) & 1;
  const int r    = row0 + (lane & 15);
  const unsigned short* p = base + (size_t)r * (size_t)ld + k0 + half * 8;
  FragU u;
  u.h[0] = *(const us8*)p;
  u.h[1] = *(const us8*)(p + 16);
  return u.f;
}

// B-operand fragment, 32x16 bf16, loaded from B^T stored row-major [N][K]
// with leading dim `ld`.  Lanes 0-15 col N=lane hold K = {0..7} u {16..23},
// lanes 16-31 hold K = {8..15} u {24..31} — two 16B loads per lane.
__device__ __forceinline__ v16bf load_fragB(const unsigned short* baseT, int ld,
                                            int n0, int k0, int lane) {
  const int half = (lane >> 4) & 1;
  const int c    = n0 + (lane & 15);
  const unsigned short* p = baseT + (size_t)c * (size_t)ld + k0 + half * 8;
  FragU u;
  u.h[0] = *(const us8*)p;
  u.h[1] = *(const us8*)(p + 16);
  return u.f;
}

// ---------------------------------------------------------------------------
// Elementwise f32 -> bf16 convert.
__global__ void __launch_bounds__(256) cvt_bf16_kernel(
    const float* __restrict__ in, unsigned short* __restrict__ out, int n) {
  int i = blockIdx.x * blockDim.x + threadIdx.x;
  if (i < n) out[i] = f2bf(in[i]);
}

// Transpose-convert a DxD fp32 weight into bf16 [N][K] (= W^T row-major).
__global__ void __launch_bounds__(256) wtrans_bf16_kernel(
    const float* __restrict__ W, unsigned short* __restrict__ Wt) {
  int i = blockIdx.x * blockDim.x + threadIdx.x;   // i = k*D + n, coalesced read
  int k = i >> 10;
  int n = i & (D_ - 1);
  Wt[(size_t)n * D_ + k] = f2bf(W[i]);
}

// ---------------------------------------------------------------------------
// bf16 WMMA GEMM: C[m,n] = sum_k A[m,k] * Bt[n,k] + bias[n]
// One wave computes a 32x64 strip: 2 A-frags reuse 4 B-frags -> 8 WMMA/k-step.
// mode: 0=Q, 1=K (both -> [B*H,S,HD] bf16), 2=V (-> [B*H,HD,S] bf16),
// 3=O (-> fp32 row-major d_out).
__global__ void __launch_bounds__(128) gemm_bf16_wmma(
    const unsigned short* __restrict__ A,   // [M x K] bf16
    const unsigned short* __restrict__ Bt,  // [N x K] bf16 (B transposed)
    const float* __restrict__ bias,         // [N] fp32
    void* __restrict__ out,
    int M, int N, int K, int mode) {
  const int lane   = threadIdx.x & 31;
  const int wave   = blockIdx.x * (blockDim.x >> 5) + (threadIdx.x >> 5);
  const int mTiles = M >> 5;               // 32-row tiles
  const int mt = wave % mTiles;
  const int nt = wave / mTiles;
  const int m0 = mt << 5;
  const int n0 = nt << 6;
  if (n0 >= N) return;   // uniform per wave: EXEC stays all-ones for WMMA

  v8f acc[8];            // [sub(2)][group(4)]
  v8f z = {};
  for (int i = 0; i < 8; ++i) acc[i] = z;

  for (int k0 = 0; k0 < K; k0 += 32) {
    v16bf a0 = load_fragA(A, K, m0,      k0, lane);
    v16bf a1 = load_fragA(A, K, m0 + 16, k0, lane);
    v16bf b0 = load_fragB(Bt, K, n0 +  0, k0, lane);
    v16bf b1 = load_fragB(Bt, K, n0 + 16, k0, lane);
    v16bf b2 = load_fragB(Bt, K, n0 + 32, k0, lane);
    v16bf b3 = load_fragB(Bt, K, n0 + 48, k0, lane);
    acc[0] = wmma_bf16(a0, b0, acc[0]);
    acc[1] = wmma_bf16(a0, b1, acc[1]);
    acc[2] = wmma_bf16(a0, b2, acc[2]);
    acc[3] = wmma_bf16(a0, b3, acc[3]);
    acc[4] = wmma_bf16(a1, b0, acc[4]);
    acc[5] = wmma_bf16(a1, b1, acc[5]);
    acc[6] = wmma_bf16(a1, b2, acc[6]);
    acc[7] = wmma_bf16(a1, b3, acc[7]);
  }

  const int half = (lane >> 4) & 1;
  const int nl   = lane & 15;
  for (int sub = 0; sub < 2; ++sub) {
    for (int g = 0; g < 4; ++g) {
      const int n  = n0 + g * 16 + nl;
      const float bv = bias[n];
      const v8f a = acc[sub * 4 + g];
      for (int v = 0; v < 8; ++v) {
        const int   m   = m0 + sub * 16 + v + 8 * half;
        const float val = a[v] + bv;
        if (mode == 3) {
          ((float*)out)[(size_t)m * N + n] = val;
        } else {
          const int b  = m >> 11;          // m / S_
          const int s  = m & (S_ - 1);
          const int h  = n >> 6;           // n / HD_
          const int hd = n & (HD_ - 1);
          const unsigned short bfv = f2bf(val);
          if (mode == 2) {  // V transposed: [B*H, HD, S]
            ((unsigned short*)out)[(((size_t)(b * H_ + h) * HD_) + hd) * S_ + s] = bfv;
          } else {          // Q/K: [B*H, S, HD]
            ((unsigned short*)out)[(((size_t)(b * H_ + h) * S_) + s) * HD_ + hd] = bfv;
          }
        }
      }
    }
  }
}

// ---------------------------------------------------------------------------
// Online-softmax update for one 16x16x2 score tile pair held in C-layout.
// Row M = v + 8*half lives across the 16 lanes of this half-wave; xor-shuffle
// reductions preserve bit4 so they stay within the half.
__device__ __forceinline__ void softmax_tile(
    const v8f& c0, const v8f& c1, float* mrow, float* lrow, v8f* acc,
    unsigned short* Pl, int half, int nl, float scale) {
  for (int v = 0; v < 8; ++v) {
    float s0 = c0[v] * scale;
    float s1 = c1[v] * scale;
    float x = fmaxf(s0, s1);
    x = fmaxf(x, __shfl_xor(x, 1));
    x = fmaxf(x, __shfl_xor(x, 2));
    x = fmaxf(x, __shfl_xor(x, 4));
    x = fmaxf(x, __shfl_xor(x, 8));
    const float mnew = fmaxf(mrow[v], x);
    const float corr = __expf(mrow[v] - mnew);
    const float p0 = __expf(s0 - mnew);
    const float p1 = __expf(s1 - mnew);
    float ps = p0 + p1;
    ps += __shfl_xor(ps, 1);
    ps += __shfl_xor(ps, 2);
    ps += __shfl_xor(ps, 4);
    ps += __shfl_xor(ps, 8);
    lrow[v] = lrow[v] * corr + ps;
    mrow[v] = mnew;
    acc[0][v] *= corr;
    acc[1][v] *= corr;
    acc[2][v] *= corr;
    acc[3][v] *= corr;
    const int r = v + 8 * half;
    Pl[r * 32 + nl]      = f2bf(p0);   // C-layout -> row-major 16x32 in LDS
    Pl[r * 32 + 16 + nl] = f2bf(p1);
  }
}

// ---------------------------------------------------------------------------
// Flash attention: one wave per (b,h, 32-row q tile pair). Streams 32
// keys/iter; both q-tiles reuse every K/V fragment (16 WMMA per iteration).
// Q,K in [B*H, S, HD] bf16; V transposed in [B*H, HD, S] bf16.
// Output (heads re-merged) bf16 [B, S, D].
__global__ void __launch_bounds__(128) attn_flash_wmma(
    const unsigned short* __restrict__ Q,
    const unsigned short* __restrict__ Km,
    const unsigned short* __restrict__ Vt,
    unsigned short* __restrict__ Oatt) {
  __shared__ unsigned short ldsP[4][32 * 32];   // per-wave P staging (2 KB each)

  const int lane  = threadIdx.x & 31;
  const int wslot = threadIdx.x >> 5;
  const int wave  = blockIdx.x * 4 + wslot;
  const int bh = wave >> 6;        // / (S_/32)
  const int qt = wave & 63;
  const int q0 = qt << 5;          // 32 query rows

  const unsigned short* Qb = Q  + (size_t)bh * S_ * HD_;
  const unsigned short* Kb = Km + (size_t)bh * S_ * HD_;
  const unsigned short* Vb = Vt + (size_t)bh * HD_ * S_;

  // Q tile 32x64 held in registers as four A fragments.
  const v16bf qa00 = load_fragA(Qb, HD_, q0,       0, lane);
  const v16bf qa01 = load_fragA(Qb, HD_, q0,      32, lane);
  const v16bf qa10 = load_fragA(Qb, HD_, q0 + 16,  0, lane);
  const v16bf qa11 = load_fragA(Qb, HD_, q0 + 16, 32, lane);

  const int half = (lane >> 4) & 1;
  const int nl   = lane & 15;
  const float scale = 0.125f;      // 1/sqrt(HD)

  float mrowA[8], lrowA[8], mrowB[8], lrowB[8];
  v8f accA[4], accB[4];
  v8f z = {};
  for (int v = 0; v < 8; ++v) {
    mrowA[v] = -1e30f; lrowA[v] = 0.0f;
    mrowB[v] = -1e30f; lrowB[v] = 0.0f;
  }
  for (int g = 0; g < 4; ++g) { accA[g] = z; accB[g] = z; }

  unsigned short* PlA = &ldsP[wslot][0];          // rows 0..15
  unsigned short* PlB = &ldsP[wslot][16 * 32];    // rows 16..31

  for (int kk = 0; kk < S_; kk += 32) {
    // Scores: two 16x32 score strips share the four K fragments.
    v16bf b00 = load_fragB(Kb, HD_, kk,       0, lane);
    v16bf b01 = load_fragB(Kb, HD_, kk,      32, lane);
    v16bf b10 = load_fragB(Kb, HD_, kk + 16,  0, lane);
    v16bf b11 = load_fragB(Kb, HD_, kk + 16, 32, lane);

    v8f cA0 = z, cA1 = z, cB0 = z, cB1 = z;
    cA0 = wmma_bf16(qa00, b00, cA0);
    cA0 = wmma_bf16(qa01, b01, cA0);
    cA1 = wmma_bf16(qa00, b10, cA1);
    cA1 = wmma_bf16(qa01, b11, cA1);
    cB0 = wmma_bf16(qa10, b00, cB0);
    cB0 = wmma_bf16(qa11, b01, cB0);
    cB1 = wmma_bf16(qa10, b10, cB1);
    cB1 = wmma_bf16(qa11, b11, cB1);

    softmax_tile(cA0, cA1, mrowA, lrowA, accA, PlA, half, nl, scale);
    softmax_tile(cB0, cB1, mrowB, lrowB, accB, PlB, half, nl, scale);

    // Reload P in A-fragment layout (in-wave LDS, DS ops are in-order).
    v16bf paA = load_fragA(PlA, 32, 0, 0, lane);
    v16bf paB = load_fragA(PlB, 32, 0, 0, lane);

    // acc += P (16x32) x V (32x64); each V fragment feeds both q-tiles.
    for (int g = 0; g < 4; ++g) {
      v16bf vb = load_fragB(Vb, S_, g * 16, kk, lane);
      accA[g] = wmma_bf16(paA, vb, accA[g]);
      accB[g] = wmma_bf16(paB, vb, accB[g]);
    }
  }

  // Normalize and write merged-head bf16 output [B, S, D].
  const int b = bh >> 4;
  const int h = bh & (H_ - 1);
  for (int v = 0; v < 8; ++v) {
    const float invA = 1.0f / lrowA[v];
    const float invB = 1.0f / lrowB[v];
    const int sA = q0 + v + 8 * half;
    const int sB = sA + 16;
    const size_t rowA = ((size_t)b * S_ + sA) * D_ + h * HD_;
    const size_t rowB = ((size_t)b * S_ + sB) * D_ + h * HD_;
    for (int g = 0; g < 4; ++g) {
      Oatt[rowA + g * 16 + nl] = f2bf(accA[g][v] * invA);
      Oatt[rowB + g * 16 + nl] = f2bf(accB[g][v] * invB);
    }
  }
}

// ---------------------------------------------------------------------------
extern "C" void kernel_launch(void* const* d_in, const int* in_sizes, int n_in,
                              void* d_out, int out_size, void* d_ws, size_t ws_size,
                              hipStream_t stream) {
  const float* x  = (const float*)d_in[0];
  const float* Wq = (const float*)d_in[1];
  const float* bq = (const float*)d_in[2];
  const float* Wk = (const float*)d_in[3];
  const float* bk = (const float*)d_in[4];
  const float* Wv = (const float*)d_in[5];
  const float* bv = (const float*)d_in[6];
  const float* Wo = (const float*)d_in[7];
  const float* bo = (const float*)d_in[8];
  float* out = (float*)d_out;

  // Workspace layout (bytes).
  const size_t MB = 1024ull * 1024ull;
  char* ws = (char*)d_ws;
  unsigned short* xb   = (unsigned short*)(ws + 0);        //  8 MB  x bf16
  unsigned short* wqT  = (unsigned short*)(ws + 8 * MB);   //  2 MB
  unsigned short* wkT  = (unsigned short*)(ws + 10 * MB);  //  2 MB
  unsigned short* wvT  = (unsigned short*)(ws + 12 * MB);  //  2 MB
  unsigned short* woT  = (unsigned short*)(ws + 14 * MB);  //  2 MB
  unsigned short* Qws  = (unsigned short*)(ws + 16 * MB);  //  8 MB  [B*H,S,HD]
  unsigned short* Kws  = (unsigned short*)(ws + 24 * MB);  //  8 MB  [B*H,S,HD]
  unsigned short* VtWs = (unsigned short*)(ws + 32 * MB);  //  8 MB  [B*H,HD,S]
  unsigned short* AttW = (unsigned short*)(ws + 40 * MB);  //  8 MB  [B,S,D]

  // 1) Precision conversion + weight transposes.
  cvt_bf16_kernel<<<(M_ * D_) / 256, 256, 0, stream>>>(x, xb, M_ * D_);
  wtrans_bf16_kernel<<<(D_ * D_) / 256, 256, 0, stream>>>(Wq, wqT);
  wtrans_bf16_kernel<<<(D_ * D_) / 256, 256, 0, stream>>>(Wk, wkT);
  wtrans_bf16_kernel<<<(D_ * D_) / 256, 256, 0, stream>>>(Wv, wvT);
  wtrans_bf16_kernel<<<(D_ * D_) / 256, 256, 0, stream>>>(Wo, woT);

  // 2) QKV projections: (M/32)*(N/64) = 2048 waves -> 512 blocks of 4 waves.
  const int gemmBlocks = (M_ / 32) * (D_ / 64) / 4;
  gemm_bf16_wmma<<<gemmBlocks, 128, 0, stream>>>(xb, wqT, bq, Qws,  M_, D_, D_, 0);
  gemm_bf16_wmma<<<gemmBlocks, 128, 0, stream>>>(xb, wkT, bk, Kws,  M_, D_, D_, 1);
  gemm_bf16_wmma<<<gemmBlocks, 128, 0, stream>>>(xb, wvT, bv, VtWs, M_, D_, D_, 2);

  // 3) Flash attention: B*H*(S/32) = 2048 waves -> 512 blocks of 4 waves.
  attn_flash_wmma<<<(B_ * H_ * (S_ / 32)) / 4, 128, 0, stream>>>(Qws, Kws, VtWs, AttW);

  // 4) Output projection -> fp32 d_out.
  gemm_bf16_wmma<<<gemmBlocks, 128, 0, stream>>>(AttW, woT, bo, out, M_, D_, D_, 3);
}
